// MultiHeadAttention_4801773437421
// MI455X (gfx1250) — compile-verified, matrix operand order checked by reading, not run on hardware
//
#include <hip/hip_runtime.h>

// ---------------------------------------------------------------------------
// Types
// ---------------------------------------------------------------------------
typedef __bf16 bf16;
typedef __attribute__((ext_vector_type(16))) __bf16 v16bf;
typedef __attribute__((ext_vector_type(8)))  __bf16 v8bf;
typedef __attribute__((ext_vector_type(4)))  __bf16 v4bf;
typedef __attribute__((ext_vector_type(8)))  float  v8f;

// Problem constants (match reference)
#define B_   2
#define S_   2048
#define DM   2048
#define NH   16
#define HD   128
#define BS_  (B_ * S_)          // 4096 rows
#define QKV3 (3 * DM)           // 6144

// ---------------------------------------------------------------------------
// CDNA5 async global->LDS copy (ASYNCcnt path), guarded so the file compiles
// on toolchains that don't expose the builtin.
// Probe-derived signature: arg0 = global int4* (b128 payload), arg1 = LDS ptr.
// ---------------------------------------------------------------------------
#if defined(__has_builtin)
#  if __has_builtin(__builtin_amdgcn_global_load_async_to_lds_b128)
#    define HAVE_ASYNC_COPY 1
#  endif
#endif
#ifndef HAVE_ASYNC_COPY
#  define HAVE_ASYNC_COPY 0
#endif

typedef int v4i __attribute__((vector_size(4 * sizeof(int))));
typedef __attribute__((address_space(1))) v4i* as1_v4i_p;
typedef __attribute__((address_space(3))) v4i* as3_v4i_p;

__device__ __forceinline__ void copy16_g2l(const bf16* __restrict__ g,
                                           bf16* __restrict__ l) {
#if HAVE_ASYNC_COPY
    __builtin_amdgcn_global_load_async_to_lds_b128(
        (as1_v4i_p)(g), (as3_v4i_p)(l), /*offset=*/0, /*cpol=*/0);
#else
    *(v8bf*)l = *(const v8bf*)g;
#endif
}

__device__ __forceinline__ void async_join() {
#if HAVE_ASYNC_COPY
    asm volatile("s_wait_asynccnt 0" ::: "memory");
#endif
}

// ---------------------------------------------------------------------------
// WMMA helpers
// ---------------------------------------------------------------------------
__device__ __forceinline__ v8f wmma_bf16(v16bf a, v16bf b, v8f c) {
    // v_wmma_f32_16x16x32_bf16  D = A(16x32) * B(32x16) + C
    return __builtin_amdgcn_wmma_f32_16x16x32_bf16(
        /*neg_a=*/false, a, /*neg_b=*/false, b,
        /*c_mod=*/(short)0, c, /*reuse_a=*/false, /*reuse_b=*/false);
}

// Load one 16x32 (A-style) or 32x16-as-columns (B-style) bf16 fragment.
// ISA layout: lanes 0-15 hold K={0..7,16..23}, lanes 16-31 hold K={8..15,24..31},
// row/column index = lane & 15.  Two 16-byte loads per lane.
__device__ __forceinline__ v16bf load_frag(const bf16* base, int stride, int lane) {
    const int row = lane & 15;
    const int kb  = (lane >> 4) << 3;       // 0 or 8
    const bf16* p = base + row * stride + kb;
    v8bf lo = *(const v8bf*)(p);
    v8bf hi = *(const v8bf*)(p + 16);
    return __builtin_shufflevector(lo, hi, 0,1,2,3,4,5,6,7,8,9,10,11,12,13,14,15);
}

// ---------------------------------------------------------------------------
// fp32 -> bf16 conversion (reference casts all inputs to bf16, RNE)
// ---------------------------------------------------------------------------
__global__ void cvt_f32_to_bf16(const float* __restrict__ in,
                                bf16* __restrict__ out, int n4) {
    int i = blockIdx.x * blockDim.x + threadIdx.x;
    if (i >= n4) return;
    const float4 f = *(const float4*)(in + (size_t)i * 4);
    v4bf o;
    o[0] = (bf16)f.x; o[1] = (bf16)f.y; o[2] = (bf16)f.z; o[3] = (bf16)f.w;
    *(v4bf*)(out + (size_t)i * 4) = o;
}

// ---------------------------------------------------------------------------
// Tiled bf16 GEMM:  C[M,N] = A[M,K] * B[K,N]   (row-major, bf16 out)
// Block tile 128x128, K-step 64.  8 waves in 2(M) x 4(N); each wave owns
// 64x32 of C (4x2 WMMA accum tiles) -> 16 WMMAs per barrier pair.
// ---------------------------------------------------------------------------
#define GBM 128
#define GBN 128
#define GBK 64

__global__ __launch_bounds__(256)
void gemm_bf16_wmma(const bf16* __restrict__ A, const bf16* __restrict__ Bm,
                    bf16* __restrict__ C, int M, int N, int K) {
    __shared__ __align__(16) bf16 As[GBM * GBK];    // 16 KB, row-major (stride 64)
    __shared__ __align__(16) bf16 BsT[GBN * GBK];   // 16 KB, n-major   (stride 64)

    const int tid  = threadIdx.x;
    const int lane = tid & 31;
    const int wid  = tid >> 5;
    const int wm   = wid >> 2;          // 0..1  (64-row slab)
    const int wn   = wid & 3;           // 0..3  (32-col slab)
    const int mBase = blockIdx.y * GBM;
    const int nBase = blockIdx.x * GBN;

    const v8f zf = {0.f,0.f,0.f,0.f,0.f,0.f,0.f,0.f};
    v8f acc[4][2];
#pragma unroll
    for (int mi = 0; mi < 4; ++mi)
#pragma unroll
        for (int ni = 0; ni < 2; ++ni) acc[mi][ni] = zf;

    for (int kb = 0; kb < K; kb += GBK) {
        // ---- prefetch next A K-slab (global_prefetch_b8) ----
        if (kb + GBK < K) {
            const int row = tid >> 1;   // 0..127
            __builtin_prefetch(&A[(size_t)(mBase + row) * K + kb + GBK +
                                  (tid & 1) * 32], 0, 1);
        }
        // ---- stage A tile (128x64): 1024 x 16B chunks, 4/thread (async) ----
#pragma unroll
        for (int c = 0; c < 4; ++c) {
            const int chunk = tid + c * 256;
            const int row   = chunk >> 3;            // 0..127
            const int col8  = (chunk & 7) * 8;       // 0..56
            copy16_g2l(&A[(size_t)(mBase + row) * K + kb + col8],
                       &As[row * GBK + col8]);
        }
        // ---- stage B tile transposed (64x128 -> BsT[n][k]) ----
#pragma unroll
        for (int c = 0; c < 4; ++c) {
            const int chunk = tid + c * 256;
            const int kk = chunk >> 4;               // 0..63
            const int n8 = (chunk & 15) * 8;         // 0..120
            const v8bf bv =
                *(const v8bf*)&Bm[(size_t)(kb + kk) * N + nBase + n8];
#pragma unroll
            for (int i = 0; i < 8; ++i) BsT[(n8 + i) * GBK + kk] = bv[i];
        }
        async_join();
        __syncthreads();

#pragma unroll
        for (int kc = 0; kc < 2; ++kc) {
            v16bf af[4], bf_[2];
#pragma unroll
            for (int mi = 0; mi < 4; ++mi)
                af[mi] = load_frag(&As[(wm * 64 + mi * 16) * GBK + kc * 32],
                                   GBK, lane);
#pragma unroll
            for (int ni = 0; ni < 2; ++ni)
                bf_[ni] = load_frag(&BsT[(wn * 32 + ni * 16) * GBK + kc * 32],
                                    GBK, lane);
#pragma unroll
            for (int mi = 0; mi < 4; ++mi)
#pragma unroll
                for (int ni = 0; ni < 2; ++ni)
                    acc[mi][ni] = wmma_bf16(af[mi], bf_[ni], acc[mi][ni]);
        }
        __syncthreads();
    }

    // ---- epilogue: C layout lane(0-15)->M=r, lane(16-31)->M=r+8, col=lane&15
    const int hi16 = (lane >> 4) ? 8 : 0;
    const int ln   = lane & 15;
#pragma unroll
    for (int mi = 0; mi < 4; ++mi)
#pragma unroll
        for (int ni = 0; ni < 2; ++ni) {
            const int row0 = mBase + wm * 64 + mi * 16 + hi16;
            const int col  = nBase + wn * 32 + ni * 16 + ln;
#pragma unroll
            for (int r = 0; r < 8; ++r)
                C[(size_t)(row0 + r) * N + col] = (bf16)acc[mi][ni][r];
        }
}

// ---------------------------------------------------------------------------
// RoPE + head split: qkv[BS,6144] -> q,k (rotated), v in [B,H,S,128] bf16
// One thread per (b,h,s,d<64) pair.
// ---------------------------------------------------------------------------
__global__ void rope_split(const bf16* __restrict__ qkv,
                           bf16* __restrict__ q, bf16* __restrict__ k,
                           bf16* __restrict__ v) {
    const int half = HD / 2;  // 64
    int idx = blockIdx.x * blockDim.x + threadIdx.x;
    const int d  = idx & (half - 1);
    int t        = idx >> 6;                 // /64
    const int s  = t & (S_ - 1);
    t          >>= 11;                       // /2048
    const int h  = t & (NH - 1);
    const int b  = t >> 4;
    if (b >= B_) return;

    const size_t src = ((size_t)(b * S_ + s)) * QKV3 + h * HD;
    const size_t dst = ((size_t)((b * NH + h) * S_ + s)) * HD;

    const float inv = __powf(10000.0f, -(float)d / (float)half);
    const float ang = (float)s * inv;
    float sn, cs;
    __sincosf(ang, &sn, &cs);

    // q
    {
        const float x1 = (float)qkv[src + d];
        const float x2 = (float)qkv[src + d + half];
        q[dst + d]        = (bf16)(x1 * cs - x2 * sn);
        q[dst + d + half] = (bf16)(x1 * sn + x2 * cs);
    }
    // k
    {
        const float x1 = (float)qkv[src + DM + d];
        const float x2 = (float)qkv[src + DM + d + half];
        k[dst + d]        = (bf16)(x1 * cs - x2 * sn);
        k[dst + d + half] = (bf16)(x1 * sn + x2 * cs);
    }
    // v (plain copy into [B,H,S,D])
    v[dst + d]        = qkv[src + 2 * DM + d];
    v[dst + d + half] = qkv[src + 2 * DM + d + half];
}

// ---------------------------------------------------------------------------
// Flash attention (causal, online softmax), bf16 WMMA.
// One block = 128 query rows of one (b,h). 8 waves, each owns 16 rows.
// Key tiles of 32; K staged row-major in LDS (async), V staged transposed.
// Output written as [B, S, H*D] so the out-projection GEMM reads it directly.
// ---------------------------------------------------------------------------
#define QT 128
#define KT 32

__global__ __launch_bounds__(256)
void attn_fwd(const bf16* __restrict__ q, const bf16* __restrict__ k,
              const bf16* __restrict__ v, bf16* __restrict__ out) {
    __shared__ __align__(16) bf16 Ks[KT * HD];      // 32 x 128  (8 KB)
    __shared__ __align__(16) bf16 VsT[HD * KT];     // 128 x 32  (8 KB)
    __shared__ __align__(16) bf16 Ps[8][16 * KT];   // per-wave P (8 KB)

    const int tid  = threadIdx.x;
    const int lane = tid & 31;
    const int wid  = tid >> 5;
    const int ln   = lane & 15;
    const int hi16 = (lane >> 4) ? 8 : 0;

    const int blk   = blockIdx.x;
    const int qTile = blk & (S_ / QT - 1);
    const int bh    = blk / (S_ / QT);
    const int h     = bh & (NH - 1);
    const int b     = bh >> 4;
    const size_t base = (size_t)bh * S_ * HD;

    const int qRow0   = qTile * QT + wid * 16;
    const float scale = 0.08838834764831845f;   // 1/sqrt(128)

    // Q fragments for this wave's 16 rows (16 x 128 = 4 chunks of 32)
    v16bf qf[4];
#pragma unroll
    for (int kc = 0; kc < 4; ++kc)
        qf[kc] = load_frag(q + base + (size_t)qRow0 * HD + kc * 32, HD, lane);

    const v8f zf = {0.f,0.f,0.f,0.f,0.f,0.f,0.f,0.f};
    v8f o[8];
#pragma unroll
    for (int nt = 0; nt < 8; ++nt) o[nt] = zf;
    float rmax[8], rsum[8];
#pragma unroll
    for (int r = 0; r < 8; ++r) { rmax[r] = -1e30f; rsum[r] = 0.f; }

    const int nTiles = (qTile * QT + QT) / KT;   // causal bound for the block
    for (int kt = 0; kt < nTiles; ++kt) {
        const int kBase = kt * KT;
        __syncthreads();
        // ---- stage K tile (32x128, async) + V tile transposed (128x32) ----
#pragma unroll
        for (int c = 0; c < 2; ++c) {
            const int chunk = tid + c * 256;        // 512 x 16B
            const int row   = chunk >> 4;
            const int col8  = (chunk & 15) * 8;
            copy16_g2l(&k[base + (size_t)(kBase + row) * HD + col8],
                       &Ks[row * HD + col8]);
            const v8bf vv =
                *(const v8bf*)&v[base + (size_t)(kBase + row) * HD + col8];
#pragma unroll
            for (int i = 0; i < 8; ++i) VsT[(col8 + i) * KT + row] = vv[i];
        }
        async_join();
        __syncthreads();

        // ---- logits: Q(16x128) . K^T(128x32) -> two 16x16 tiles ----
        v8f accL = zf, accH = zf;
#pragma unroll
        for (int kc = 0; kc < 4; ++kc) {
            const v16bf b0 = load_frag(&Ks[0  * HD + kc * 32], HD, lane);
            const v16bf b1 = load_frag(&Ks[16 * HD + kc * 32], HD, lane);
            accL = wmma_bf16(qf[kc], b0, accL);
            accH = wmma_bf16(qf[kc], b1, accH);
        }

        // ---- online softmax (rows split across half-waves) ----
        float p0[8], p1[8], tmax[8];
#pragma unroll
        for (int r = 0; r < 8; ++r) {
            const int qAbs = qRow0 + hi16 + r;
            float l0 = accL[r] * scale;
            float l1 = accH[r] * scale;
            if (kBase + ln      > qAbs) l0 = -1e30f;
            if (kBase + 16 + ln > qAbs) l1 = -1e30f;
            p0[r] = l0; p1[r] = l1;
            tmax[r] = fmaxf(l0, l1);
        }
#pragma unroll
        for (int r = 0; r < 8; ++r)
#pragma unroll
            for (int m = 1; m < 16; m <<= 1)
                tmax[r] = fmaxf(tmax[r], __shfl_xor(tmax[r], m, 32));
#pragma unroll
        for (int r = 0; r < 8; ++r) {
            const float nm   = fmaxf(rmax[r], tmax[r]);
            const float corr = __expf(rmax[r] - nm);
            rmax[r] = nm;
            p0[r] = __expf(p0[r] - nm);
            p1[r] = __expf(p1[r] - nm);
            float ts = p0[r] + p1[r];
#pragma unroll
            for (int m = 1; m < 16; m <<= 1) ts += __shfl_xor(ts, m, 32);
            rsum[r] = rsum[r] * corr + ts;
#pragma unroll
            for (int nt = 0; nt < 8; ++nt) o[nt][r] *= corr;
        }

        // ---- P through per-wave LDS scratch to re-swizzle into A layout ----
#pragma unroll
        for (int r = 0; r < 8; ++r) {
            const int row = hi16 + r;
            Ps[wid][row * KT + ln]      = (bf16)p0[r];
            Ps[wid][row * KT + 16 + ln] = (bf16)p1[r];
        }
        asm volatile("s_wait_dscnt 0" ::: "memory");  // wave-local LDS RAW
        const v16bf pa = load_frag(&Ps[wid][0], KT, lane);

        // ---- O += P(16x32) . V(32x128) ----
#pragma unroll
        for (int nt = 0; nt < 8; ++nt) {
            const v16bf vb = load_frag(&VsT[nt * 16 * KT], KT, lane);
            o[nt] = wmma_bf16(pa, vb, o[nt]);
        }
    }

    // ---- epilogue: normalize, write [B, S, H*D] bf16 ----
#pragma unroll
    for (int nt = 0; nt < 8; ++nt) {
        const int dcol = h * HD + nt * 16 + ln;
#pragma unroll
        for (int r = 0; r < 8; ++r) {
            const int rowAbs = qRow0 + hi16 + r;
            out[(size_t)(b * S_ + rowAbs) * DM + dcol] =
                (bf16)(o[nt][r] / rsum[r]);
        }
    }
}

// ---------------------------------------------------------------------------
// Host-side launcher (graph-capture safe: only kernel launches on `stream`)
// ---------------------------------------------------------------------------
extern "C" void kernel_launch(void* const* d_in, const int* in_sizes, int n_in,
                              void* d_out, int out_size, void* d_ws, size_t ws_size,
                              hipStream_t stream) {
    (void)in_sizes; (void)n_in; (void)out_size; (void)ws_size;
    const float* x      = (const float*)d_in[0];   // [B,S,DM] f32
    const float* w_qkv  = (const float*)d_in[1];   // [DM,3DM] f32
    const float* w_out  = (const float*)d_in[2];   // [DM,DM]  f32
    bf16* out = (bf16*)d_out;                      // [B,S,DM] bf16

    char* ws = (char*)d_ws;
    size_t off = 0;
    auto alloc = [&](size_t elems) {
        void* p = ws + off;
        off += (elems * sizeof(bf16) + 255) & ~(size_t)255;
        return (bf16*)p;
    };
    bf16* xb    = alloc((size_t)BS_ * DM);        // x bf16
    bf16* wqb   = alloc((size_t)DM * QKV3);       // w_qkv bf16
    bf16* wob   = alloc((size_t)DM * DM);         // w_out bf16
    bf16* qkv   = alloc((size_t)BS_ * QKV3);      // qkv projection
    bf16* qh    = alloc((size_t)B_ * NH * S_ * HD);
    bf16* kh    = alloc((size_t)B_ * NH * S_ * HD);
    bf16* vh    = alloc((size_t)B_ * NH * S_ * HD);
    bf16* attn  = alloc((size_t)BS_ * DM);        // attention out [BS, DM]

    // 1) convert inputs to bf16 (4 elems/thread)
    {
        int n4 = (BS_ * DM) / 4;
        cvt_f32_to_bf16<<<(n4 + 255) / 256, 256, 0, stream>>>(x, xb, n4);
        n4 = (DM * QKV3) / 4;
        cvt_f32_to_bf16<<<(n4 + 255) / 256, 256, 0, stream>>>(w_qkv, wqb, n4);
        n4 = (DM * DM) / 4;
        cvt_f32_to_bf16<<<(n4 + 255) / 256, 256, 0, stream>>>(w_out, wob, n4);
    }

    // 2) QKV projection: [4096,2048] x [2048,6144]
    {
        dim3 grid(QKV3 / GBN, BS_ / GBM);
        gemm_bf16_wmma<<<grid, 256, 0, stream>>>(xb, wqb, qkv, BS_, QKV3, DM);
    }

    // 3) RoPE + head split
    {
        const int total = B_ * NH * S_ * (HD / 2);
        rope_split<<<total / 256, 256, 0, stream>>>(qkv, qh, kh, vh);
    }

    // 4) causal flash attention
    {
        const int grid = B_ * NH * (S_ / QT);
        attn_fwd<<<grid, 256, 0, stream>>>(qh, kh, vh, attn);
    }

    // 5) output projection: [4096,2048] x [2048,2048] -> d_out (bf16)
    {
        dim3 grid(DM / GBN, BS_ / GBM);
        gemm_bf16_wmma<<<grid, 256, 0, stream>>>(attn, wob, out, BS_, DM, DM);
    }
}
